// FusedMoE_39238821216260
// MI455X (gfx1250) — matrix operand order repaired and checked by reading, not compiled, not run on hardware
//
#include <hip/hip_runtime.h>
#include <hip/hip_bf16.h>

// Problem constants (match reference)
#define T_TOK 4096
#define H_DIM 2048
#define I_DIM 4096
#define E_EXP 8

// Tiling
#define M_BLK 64           // tokens per workgroup tile
#define N_BLK 128          // output columns per workgroup
#define KC    64           // K-chunk staged in LDS (two 16x16x32 WMMA steps)
#define XS_ST (KC + 8)     // LDS row stride for A tiles (keeps 16B alignment)
#define WS_ST (KC + 16)    // LDS row stride for B tiles (keeps 32B alignment)

typedef __attribute__((ext_vector_type(16))) __bf16          v16bf;
typedef __attribute__((ext_vector_type(8)))  float           v8f;
typedef __attribute__((ext_vector_type(8)))  unsigned short  ush8;
typedef __attribute__((ext_vector_type(16))) unsigned short  ush16;
typedef __attribute__((ext_vector_type(4)))  unsigned short  ush4;
typedef __attribute__((ext_vector_type(2)))  int             v2i_t;

// CDNA5 async global->LDS path (guarded; falls back to plain copies)
#if defined(__has_builtin)
#if __has_builtin(__builtin_amdgcn_global_load_async_to_lds_b64) && \
    __has_builtin(__builtin_amdgcn_s_wait_asynccnt)
#define USE_ASYNC_LDS 1
#endif
#if __has_builtin(__builtin_amdgcn_sched_group_barrier)
#define SCHED_GROUP(mask, size, sync) __builtin_amdgcn_sched_group_barrier(mask, size, sync)
#endif
#endif
#ifndef USE_ASYNC_LDS
#define USE_ASYNC_LDS 0
#endif
#ifndef SCHED_GROUP
#define SCHED_GROUP(mask, size, sync)
#endif

// sched_group_barrier masks (LLVM AMDGPU)
#define SG_WMMA    0x010
#define SG_DSREAD  0x200

union BF16x16 {
    ush16 u;
    v16bf v;
    struct { ush8 lo; ush8 hi; } s;
};

// f32 -> bf16 round-to-nearest-even
__device__ __forceinline__ unsigned short f2bf(float f) {
    union { float f; unsigned int u; } cv; cv.f = f;
    unsigned int u = cv.u;
    unsigned int r = u + 0x7FFFu + ((u >> 16) & 1u);
    return (unsigned short)(r >> 16);
}

// A fragment (16x32 bf16), ISA 7.12.2: lane = half*16 + m; lane holds row m,
// K = {half*8 .. half*8+7} in VGPR0-3 and {16+half*8 ..} in VGPR4-7.
__device__ __forceinline__ v16bf load_frag_a(const unsigned short* row, int kk, int half) {
    BF16x16 t;
    t.s.lo = *(const ush8*)(row + kk + half * 8);
    t.s.hi = *(const ush8*)(row + kk + 16 + half * 8);
    return t.v;
}

// B fragment (32x16 bf16): lane holds column n = lane&15,
// K = {half*8 .. half*8+7} in VGPR0-3 and {16+half*8 ..} in VGPR4-7 (matches A).
__device__ __forceinline__ v16bf load_frag_b(const unsigned short* row, int kk, int half) {
    BF16x16 t;
    t.s.lo = *(const ush8*)(row + kk + half * 8);
    t.s.hi = *(const ush8*)(row + kk + 16 + half * 8);
    return t.v;
}

// ---------------------------------------------------------------------------
// Kernel 0: zero output + routing counters
// ---------------------------------------------------------------------------
__global__ void zero_out_kernel(float* __restrict__ out, int n, int* __restrict__ cnt) {
    int i = blockIdx.x * blockDim.x + threadIdx.x;
    int stride = gridDim.x * blockDim.x;
    for (; i < n; i += stride) out[i] = 0.0f;
    if (blockIdx.x == 0 && threadIdx.x < E_EXP) cnt[threadIdx.x] = 0;
}

// ---------------------------------------------------------------------------
// Kernel 1: deterministic per-expert token compaction (one block per expert).
// ---------------------------------------------------------------------------
__global__ void route_fill_kernel(const float* __restrict__ tkw,
                                  const int* __restrict__ tkid,
                                  int* __restrict__ cnt,
                                  int* __restrict__ tokL,
                                  float* __restrict__ wL) {
    const int e = blockIdx.x;
    const int tid = threadIdx.x;
    const int lane = tid & 31;
    const int wv = tid >> 5;
    __shared__ int wcnt[8];

    int base = 0;
    for (int t0 = 0; t0 < T_TOK; t0 += 256) {
        const int t = t0 + tid;
        const int id0 = tkid[t * 2 + 0];
        const int id1 = tkid[t * 2 + 1];
        const float w0 = tkw[t * 2 + 0];
        const float w1 = tkw[t * 2 + 1];
        const float inv = 1.0f / (w0 + w1);   // renormalize=True
        float we = 0.0f;
        bool m = false;
        if (id0 == e) { we += w0 * inv; m = true; }
        if (id1 == e) { we += w1 * inv; m = true; }

        unsigned long long mask = __ballot(m);        // wave32: low 32 bits
        int lanePre = __popcll(mask & ((1ull << lane) - 1ull));
        int wtot = __popcll(mask);
        if (lane == 0) wcnt[wv] = wtot;
        __syncthreads();
        int wbase = 0, tot = 0;
        #pragma unroll
        for (int i = 0; i < 8; i++) { if (i < wv) wbase += wcnt[i]; tot += wcnt[i]; }
        if (m) {
            int pos = base + wbase + lanePre;
            tokL[e * T_TOK + pos] = t;
            wL[e * T_TOK + pos] = we;
        }
        base += tot;
        __syncthreads();
    }
    if (tid == 0) cnt[e] = base;
}

// ---------------------------------------------------------------------------
// Kernel 2: exclusive scan of per-expert counts (8 values)
// ---------------------------------------------------------------------------
__global__ void scan_off_kernel(const int* __restrict__ cnt, int* __restrict__ off) {
    if (threadIdx.x == 0 && blockIdx.x == 0) {
        int a = 0;
        for (int e = 0; e < E_EXP; e++) { off[e] = a; a += cnt[e]; }
    }
}

// ---------------------------------------------------------------------------
// Kernel 3: GEMM1  act = silu(x @ G^T) * (x @ U^T), bf16 WMMA, f32 accum.
// Grid: (I/N_BLK, E).  Block: 256 threads = 8 waves laid out 4(M) x 2(N).
// ---------------------------------------------------------------------------
__global__ __launch_bounds__(256) void moe_gemm1(
    const float* __restrict__ hidden,
    const float* __restrict__ guw,
    const int* __restrict__ cnt,
    const int* __restrict__ off,
    const int* __restrict__ tokL,
    unsigned short* __restrict__ act) {

    __shared__ unsigned short Xs[M_BLK * XS_ST];
    __shared__ unsigned short Gs[N_BLK * WS_ST];
    __shared__ unsigned short Us[N_BLK * WS_ST];
    __shared__ int tk[M_BLK];

    const int e = blockIdx.y;
    const int n0 = blockIdx.x * N_BLK;
    const int C = cnt[e];
    if (C == 0) return;
    const int offe = off[e];

    const int tid = threadIdx.x;
    const int lane = tid & 31, half = lane >> 4, l16 = lane & 15;
    const int wv = tid >> 5;
    const int wm = wv & 3;       // wave row: 4 x 16 = 64 tokens
    const int wn = wv >> 2;      // wave col: 2 x 64 = 128 outputs
    const size_t gbase = (size_t)e * (2 * (size_t)I_DIM) * H_DIM;
    const size_t upofs = (size_t)I_DIM * H_DIM;

    for (int m0 = 0; m0 < C; m0 += M_BLK) {
        if (tid < M_BLK) {
            int s = m0 + tid;
            tk[tid] = tokL[e * T_TOK + (s < C ? s : C - 1)];
        }
        __syncthreads();

        v8f accg[4], accu[4];
        const v8f vzero = {0.f, 0.f, 0.f, 0.f, 0.f, 0.f, 0.f, 0.f};
        #pragma unroll
        for (int i = 0; i < 4; i++) { accg[i] = vzero; accu[i] = vzero; }

        for (int k0 = 0; k0 < H_DIM; k0 += KC) {
            // Stage X tile (64 x 64) f32 -> bf16 LDS
            #pragma unroll
            for (int i = 0; i < 4; i++) {
                int f = tid + 256 * i;            // 1024 float4 total
                int r = f >> 4, c4 = f & 15;
                const float4 v = *(const float4*)&hidden[(size_t)tk[r] * H_DIM + k0 + c4 * 4];
                unsigned short* d = &Xs[r * XS_ST + c4 * 4];
                d[0] = f2bf(v.x); d[1] = f2bf(v.y); d[2] = f2bf(v.z); d[3] = f2bf(v.w);
            }
            // Stage gate + up weight tiles (128 x 64 each) f32 -> bf16 LDS
            #pragma unroll
            for (int i = 0; i < 8; i++) {
                int f = tid + 256 * i;            // 2048 float4 total
                int r = f >> 4, c4 = f & 15;
                const float* gp = &guw[gbase + (size_t)(n0 + r) * H_DIM + k0 + c4 * 4];
                float4 g = *(const float4*)gp;
                unsigned short* dg = &Gs[r * WS_ST + c4 * 4];
                dg[0] = f2bf(g.x); dg[1] = f2bf(g.y); dg[2] = f2bf(g.z); dg[3] = f2bf(g.w);
                float4 u = *(const float4*)(gp + upofs);
                unsigned short* du = &Us[r * WS_ST + c4 * 4];
                du[0] = f2bf(u.x); du[1] = f2bf(u.y); du[2] = f2bf(u.z); du[3] = f2bf(u.w);
            }
            // Hint the next weight K-chunk into cache (global_prefetch_b8)
            if (k0 + KC < H_DIM) {
                __builtin_prefetch(&guw[gbase + (size_t)(n0 + (tid & 127)) * H_DIM + k0 + KC], 0, 3);
            }
            __syncthreads();

            #pragma unroll
            for (int kk = 0; kk < KC; kk += 32) {
                // Preload fragments; sched_group_barrier pipeline below keeps
                // DS reads ~4 fragments ahead of the consuming WMMA.
                v16bf a = load_frag_a(&Xs[(wm * 16 + l16) * XS_ST], kk, half);
                v16bf bg[4], bu[4];
                #pragma unroll
                for (int nt = 0; nt < 4; nt++) {
                    const int r = wn * 64 + nt * 16 + l16;
                    bg[nt] = load_frag_b(&Gs[r * WS_ST], kk, half);
                    bu[nt] = load_frag_b(&Us[r * WS_ST], kk, half);
                }
                #pragma unroll
                for (int nt = 0; nt < 4; nt++) {
                    accg[nt] = __builtin_amdgcn_wmma_f32_16x16x32_bf16(
                        false, a, false, bg[nt], (short)0, accg[nt], false, false);
                    accu[nt] = __builtin_amdgcn_wmma_f32_16x16x32_bf16(
                        false, a, false, bu[nt], (short)0, accu[nt], false, false);
                }
                // Pipeline: A(2) + 4 B-frags (8) up front, then WMMA with
                // 2 DS reads issued per WMMA, tail of 4 WMMAs.
                SCHED_GROUP(SG_DSREAD, 10, 0);
                SCHED_GROUP(SG_WMMA, 1, 0); SCHED_GROUP(SG_DSREAD, 2, 0);
                SCHED_GROUP(SG_WMMA, 1, 0); SCHED_GROUP(SG_DSREAD, 2, 0);
                SCHED_GROUP(SG_WMMA, 1, 0); SCHED_GROUP(SG_DSREAD, 2, 0);
                SCHED_GROUP(SG_WMMA, 1, 0); SCHED_GROUP(SG_DSREAD, 2, 0);
                SCHED_GROUP(SG_WMMA, 4, 0);
            }
            __syncthreads();
        }

        // Epilogue: silu(gate) * up  -> bf16 act workspace (packed by pair id)
        #pragma unroll
        for (int nt = 0; nt < 4; nt++) {
            #pragma unroll
            for (int j = 0; j < 8; j++) {
                const int mrow = wm * 16 + j + half * 8;   // D layout: VGPR j -> M=j(+8)
                const int slot = m0 + mrow;
                if (slot < C) {
                    float g = accg[nt][j];
                    float u = accu[nt][j];
                    float aval = (g / (1.0f + __expf(-g))) * u;
                    const int ng = n0 + wn * 64 + nt * 16 + l16;
                    act[(size_t)(offe + slot) * I_DIM + ng] = f2bf(aval);
                }
            }
        }
        __syncthreads();
    }
}

// ---------------------------------------------------------------------------
// Kernel 4: GEMM2  out[t,:] += w * (act @ down^T).  Grid: (H/N_BLK, E).
// act tile is staged with CDNA5 GLOBAL_LOAD_ASYNC_TO_LDS (overlapped with the
// VALU-heavy f32->bf16 weight staging), drained by s_wait_asynccnt.
// ---------------------------------------------------------------------------
__global__ __launch_bounds__(256) void moe_gemm2(
    const unsigned short* __restrict__ act,
    const float* __restrict__ dww,
    const int* __restrict__ cnt,
    const int* __restrict__ off,
    const int* __restrict__ tokL,
    const float* __restrict__ wL,
    float* __restrict__ out) {

    __shared__ unsigned short As[M_BLK * XS_ST];
    __shared__ unsigned short Ds[N_BLK * WS_ST];
    __shared__ int   tk[M_BLK];
    __shared__ float tw[M_BLK];

    const int e = blockIdx.y;
    const int n0 = blockIdx.x * N_BLK;     // over H
    const int C = cnt[e];
    if (C == 0) return;
    const int offe = off[e];

    const int tid = threadIdx.x;
    const int lane = tid & 31, half = lane >> 4, l16 = lane & 15;
    const int wv = tid >> 5;
    const int wm = wv & 3;
    const int wn = wv >> 2;

    for (int m0 = 0; m0 < C; m0 += M_BLK) {
        if (tid < M_BLK) {
            int s = m0 + tid;
            tk[tid] = (s < C) ? tokL[e * T_TOK + s] : 0;
            tw[tid] = (s < C) ? wL[e * T_TOK + s] : 0.0f;
        }
        __syncthreads();

        v8f acc[4];
        const v8f vzero = {0.f, 0.f, 0.f, 0.f, 0.f, 0.f, 0.f, 0.f};
        #pragma unroll
        for (int i = 0; i < 4; i++) acc[i] = vzero;

        for (int k0 = 0; k0 < I_DIM; k0 += KC) {
            // Stage act tile (64 x 64 bf16): async global->LDS, no VGPR bounce.
            #pragma unroll
            for (int i = 0; i < 4; i++) {
                int f = tid + 256 * i;              // 1024 x 8B total
                int r = f >> 4, c4 = f & 15;
                int rs = m0 + r; if (rs >= C) rs = C - 1;   // clamp (guarded in epilogue)
                const unsigned short* gsrc = &act[(size_t)(offe + rs) * I_DIM + k0 + c4 * 4];
                unsigned short* ldst = &As[r * XS_ST + c4 * 4];
#if USE_ASYNC_LDS
                __builtin_amdgcn_global_load_async_to_lds_b64(
                    (v2i_t*)gsrc, (v2i_t*)ldst, 0, 0);
#else
                *(ush4*)ldst = *(const ush4*)gsrc;
#endif
            }
            // Stage down-weight tile (128 x 64) f32 -> bf16 LDS (overlaps async)
            #pragma unroll
            for (int i = 0; i < 8; i++) {
                int f = tid + 256 * i;
                int r = f >> 4, c4 = f & 15;
                const float4 v = *(const float4*)&dww[((size_t)e * H_DIM + n0 + r) * I_DIM + k0 + c4 * 4];
                unsigned short* d = &Ds[r * WS_ST + c4 * 4];
                d[0] = f2bf(v.x); d[1] = f2bf(v.y); d[2] = f2bf(v.z); d[3] = f2bf(v.w);
            }
            if (k0 + KC < I_DIM) {
                __builtin_prefetch(&dww[((size_t)e * H_DIM + n0 + (tid & 127)) * I_DIM + k0 + KC], 0, 3);
            }
#if USE_ASYNC_LDS
            __builtin_amdgcn_s_wait_asynccnt(0);
#endif
            __syncthreads();

            #pragma unroll
            for (int kk = 0; kk < KC; kk += 32) {
                v16bf a = load_frag_a(&As[(wm * 16 + l16) * XS_ST], kk, half);
                v16bf b[4];
                #pragma unroll
                for (int nt = 0; nt < 4; nt++) {
                    const int r = wn * 64 + nt * 16 + l16;
                    b[nt] = load_frag_b(&Ds[r * WS_ST], kk, half);
                }
                #pragma unroll
                for (int nt = 0; nt < 4; nt++) {
                    acc[nt] = __builtin_amdgcn_wmma_f32_16x16x32_bf16(
                        false, a, false, b[nt], (short)0, acc[nt], false, false);
                }
                // Pipeline: A(2) + 2 B-frags (4) up front, then WMMA with
                // 2 DS reads per WMMA, tail of 2 WMMAs.
                SCHED_GROUP(SG_DSREAD, 6, 0);
                SCHED_GROUP(SG_WMMA, 1, 0); SCHED_GROUP(SG_DSREAD, 2, 0);
                SCHED_GROUP(SG_WMMA, 1, 0); SCHED_GROUP(SG_DSREAD, 2, 0);
                SCHED_GROUP(SG_WMMA, 2, 0);
            }
            __syncthreads();
        }

        // Epilogue: weighted scatter-add into output (global_atomic_add_f32)
        #pragma unroll
        for (int nt = 0; nt < 4; nt++) {
            #pragma unroll
            for (int j = 0; j < 8; j++) {
                const int mrow = wm * 16 + j + half * 8;
                const int slot = m0 + mrow;
                if (slot < C) {
                    const int tkn = tk[mrow];
                    const float w = tw[mrow];
                    const int ng = n0 + wn * 64 + nt * 16 + l16;
                    atomicAdd(&out[(size_t)tkn * H_DIM + ng], w * acc[nt][j]);
                }
            }
        }
        __syncthreads();
    }
}

// ---------------------------------------------------------------------------
// Host launcher
// ---------------------------------------------------------------------------
extern "C" void kernel_launch(void* const* d_in, const int* in_sizes, int n_in,
                              void* d_out, int out_size, void* d_ws, size_t ws_size,
                              hipStream_t stream) {
    const float* hidden = (const float*)d_in[0];   // [T, H] f32
    const float* tkw    = (const float*)d_in[1];   // [T, K] f32
    const int*   tkid   = (const int*)d_in[2];     // [T, K] (int32 on device; jax x64 off)
    const float* guw    = (const float*)d_in[3];   // [E, 2I, H] f32
    const float* dww    = (const float*)d_in[4];   // [E, H, I] f32
    float* out = (float*)d_out;                    // [T, H] f32

    // Workspace layout
    char* ws = (char*)d_ws;
    int*   cnt  = (int*)ws;                                  // E counters
    int*   off  = cnt + E_EXP;                               // E offsets
    int*   tokL = (int*)(ws + 256);                          // E*T token ids
    float* wL   = (float*)(ws + 256 + (size_t)E_EXP * T_TOK * 4);        // E*T weights
    unsigned short* actB = (unsigned short*)(ws + 256 + (size_t)E_EXP * T_TOK * 8); // act bf16

    // 0) zero output + counters
    zero_out_kernel<<<4096, 256, 0, stream>>>(out, T_TOK * H_DIM, cnt);
    // 1) deterministic routing compaction
    route_fill_kernel<<<E_EXP, 256, 0, stream>>>(tkw, tkid, cnt, tokL, wL);
    // 2) offsets
    scan_off_kernel<<<1, 64, 0, stream>>>(cnt, off);
    // 3) gate/up GEMM + silu*up epilogue
    dim3 g1(I_DIM / N_BLK, E_EXP);
    moe_gemm1<<<g1, 256, 0, stream>>>(hidden, guw, cnt, off, tokL, actB);
    // 4) down GEMM + weighted scatter-add
    dim3 g2(H_DIM / N_BLK, E_EXP);
    moe_gemm2<<<g2, 256, 0, stream>>>(actB, dww, cnt, off, tokL, wL, out);
}
